// DilatedAttention_51135880626345
// MI455X (gfx1250) — matrix operand order repaired and checked by reading, not compiled, not run on hardware
//
#include <hip/hip_runtime.h>

// ---------------------------------------------------------------------------
// Types for CDNA5 WMMA (wave32): V_WMMA_F32_16X16X32_BF16
// ---------------------------------------------------------------------------
typedef __bf16 v16bf __attribute__((ext_vector_type(16)));
typedef float  v8f   __attribute__((ext_vector_type(8)));

union Frag {
    v16bf v;
    uint4 q[2];
};

__device__ __forceinline__ unsigned short f2bf(float f) {
    union { float f; unsigned u; } x;
    x.f = f;
    unsigned u = x.u + 0x7FFFu + ((x.u >> 16) & 1u);  // round-to-nearest-even
    return (unsigned short)(u >> 16);
}

// A-matrix fragment (16x32 bf16, MxK): lane (lo,hi) = row lo;
// K = hi*16 + t, t=0..15 contiguous
__device__ __forceinline__ v16bf load_a_frag(const unsigned short* rowbase, int hi) {
    Frag f;
    f.q[0] = *(const uint4*)(rowbase + hi * 16);
    f.q[1] = *(const uint4*)(rowbase + hi * 16 + 8);
    return f.v;
}

// B-matrix fragment (32x16 bf16, KxN): lane (lo,hi) = column lo,
// K = hi*16 + t, t=0..15 contiguous (ptr already adjusted by hi*16)
__device__ __forceinline__ v16bf load_b_frag(const unsigned short* p) {
    Frag f;
    f.q[0] = *(const uint4*)(p);
    f.q[1] = *(const uint4*)(p + 8);
    return f.v;
}

// CDNA5 async global->LDS copy (16B per lane), tracked by ASYNCcnt.
// lds_off: 32-bit LDS byte address (low 32 bits of the generic LDS pointer,
// per ISA aperture rule).
__device__ __forceinline__ void async_copy_b128(unsigned lds_off,
                                                const unsigned short* gptr) {
    asm volatile("global_load_async_to_lds_b128 %0, %1, off"
                 :: "v"(lds_off), "v"(gptr)
                 : "memory");
}
__device__ __forceinline__ void wait_asynccnt0() {
    asm volatile("s_wait_asynccnt 0x0" ::: "memory");
}

// ---------------------------------------------------------------------------
// Dilated-mask multiplicity, closed form.
// Scale X in [0,10]: r=2^X, S=2^(X+2). Covered iff
//   (i^j) < 2^(X+2)           (same segment)        -> X >= bitlen(i^j)-2
//   i%r==h%r && j%r==h%r      (dilation phase)      -> X <= ctz((i^h)|(j^h))
// count = #covered X.
// ---------------------------------------------------------------------------
__device__ __forceinline__ int dil_count(int i, int j, int h) {
    int t1 = (i ^ h) | (j ^ h);
    int L = t1 ? __builtin_ctz(t1) : 31;
    if (L > 10) L = 10;
    int bx = i ^ j;
    int Xmin = bx ? (32 - __builtin_clz((unsigned)bx) - 2) : 0;
    if (Xmin < 0) Xmin = 0;
    int c = L - Xmin + 1;
    return c > 0 ? c : 0;
}

// ---------------------------------------------------------------------------
// fp32 -> bf16 elementwise
// ---------------------------------------------------------------------------
__global__ void f32_to_bf16_kernel(const float* __restrict__ src,
                                   unsigned short* __restrict__ dst, int n) {
    int i = blockIdx.x * blockDim.x + threadIdx.x;
    if (i < n) dst[i] = f2bf(src[i]);
}

// ---------------------------------------------------------------------------
// GEMM: C[M][N] = A[M][K](bf16) * W[N][K](bf16)^T + bias[N], fp32 out.
// Block = 128 threads (4 waves), block tile 64x64 (wave = 16 rows x 64 cols).
// The 64x32 B tile is shared by all 4 waves -> staged once per K-slab in LDS
// via double-buffered GLOBAL_LOAD_ASYNC_TO_LDS_B128 (ASYNCcnt), overlapping
// the copy of slab k+32 with the WMMAs of slab k.
// ---------------------------------------------------------------------------
__global__ __launch_bounds__(128) void gemm_bf16_wmma(
    const unsigned short* __restrict__ A,
    const unsigned short* __restrict__ W,
    const float* __restrict__ bias,
    float* __restrict__ C,
    int M, int N, int K)
{
    // B staging: 2 buffers x 64 cols x 32 K-halves (row-major, 64B rows) = 8 KB
    __shared__ unsigned short Bls[2 * 64 * 32];

    int lane = threadIdx.x & 31;
    int wave = threadIdx.x >> 5;
    int hi = lane >> 4, lo = lane & 15;
    int m0 = blockIdx.y * 64 + wave * 16;
    int n0 = blockIdx.x * 64;

    unsigned bbase = (unsigned)(uintptr_t)(void*)Bls;  // LDS byte address

    v8f acc[4] = {};

    const unsigned short* arow = A + (size_t)(m0 + lo) * K;

    // stage(buf, kk): cooperative async copy of W[n0..n0+63][kk..kk+31]
    auto stage = [&](int buf, int kk) {
#pragma unroll
        for (int it = 0; it < 2; ++it) {
            int idx = wave * 64 + it * 32 + lane;  // 0..255 chunk id
            int n = idx >> 2;                       // 0..63 (column of C)
            int c = idx & 3;                        // 16B chunk within 64B row
            const unsigned short* g = W + (size_t)(n0 + n) * K + kk + c * 8;
            unsigned loff = bbase + (unsigned)(buf * 4096 + n * 64 + c * 16);
            async_copy_b128(loff, g);
        }
    };

    // prologue: fill buffer 0
    stage(0, 0);
    wait_asynccnt0();
    __syncthreads();

    for (int k0 = 0; k0 < K; k0 += 32) {
        int cur = (k0 >> 5) & 1;
        bool more = (k0 + 32) < K;
        if (more) {
            stage(cur ^ 1, k0 + 32);                    // overlap copy w/ compute
            __builtin_prefetch(arow + k0 + 64, 0, 3);   // near-scope A prefetch
        }

        v16bf a = load_a_frag(arow + k0, hi);
        const unsigned short* bbuf = Bls + cur * 2048;
#pragma unroll
        for (int g = 0; g < 4; ++g) {
            const unsigned short* bp = bbuf + (16 * g + lo) * 32 + hi * 16;
            v16bf b = load_b_frag(bp);                  // ds_load_b128 x2
            acc[g] = __builtin_amdgcn_wmma_f32_16x16x32_bf16(
                false, a, false, b, (short)0, acc[g], false, false);
        }

        if (more) wait_asynccnt0();  // own ASYNCcnt drained before barrier
        __syncthreads();             // all waves' staging visible; reads retired
    }

#pragma unroll
    for (int g = 0; g < 4; ++g) {
        int n = n0 + 16 * g + lo;
        float bv = bias[n];
#pragma unroll
        for (int i = 0; i < 8; ++i) {
            int m = m0 + i + 8 * hi;  // C layout: VGPR i -> row i + 8*hi, lane lo -> col
            C[(size_t)m * N + n] = acc[g][i] + bv;
        }
    }
}

// ---------------------------------------------------------------------------
// RoPE (full-dim, as in reference) + pack to WMMA-friendly bf16 layouts:
//   Qb, Kb : [H][T][64]   (row = token, contiguous head channels)
//   Vt     : [1024][T]    (channel-major so PV B-fragments are contiguous in key)
// Channel pair (c, c+512) shares inv_freq[c]; h = c/64, dd = c%64 => flat = c.
// ---------------------------------------------------------------------------
__global__ __launch_bounds__(512) void rope_pack_kernel(
    const float* __restrict__ qf, const float* __restrict__ kf,
    const float* __restrict__ vf,
    unsigned short* __restrict__ Qb, unsigned short* __restrict__ Kb,
    unsigned short* __restrict__ Vt)
{
    const int T = 2048, D = 1024;
    int t = blockIdx.x;
    int c = threadIdx.x;  // 0..511
    // inv_freq = 15000^(-c/512) = exp(-(c/512)*ln(15000))
    float inv = __expf(-((float)c * (1.0f / 512.0f)) * 9.6158054800843071f);
    float pos = (float)t * inv;
    float sn, cs;
    __sincosf(pos, &sn, &cs);

    size_t i1 = (size_t)t * D + c, i2 = i1 + 512;
    float qa = qf[i1], qb = qf[i2];
    float ka = kf[i1], kb = kf[i2];
    float va = vf[i1], vb = vf[i2];

    float q1 = qa * cs - qb * sn, q2 = qb * cs + qa * sn;
    float k1 = ka * cs - kb * sn, k2 = kb * cs + ka * sn;

    int c2 = c + 512;
    size_t o1 = ((size_t)(c >> 6) * T + t) * 64 + (c & 63);
    size_t o2 = ((size_t)(c2 >> 6) * T + t) * 64 + (c2 & 63);
    Qb[o1] = f2bf(q1); Qb[o2] = f2bf(q2);
    Kb[o1] = f2bf(k1); Kb[o2] = f2bf(k2);
    Vt[(size_t)c * T + t]  = f2bf(va);
    Vt[(size_t)c2 * T + t] = f2bf(vb);
}

// ---------------------------------------------------------------------------
// Flash-style dilated attention. One wave per (head, 16-query block).
// Streams keys in 32-wide blocks; QK^T and PV via bf16 WMMA; online softmax
// with -1e30 sentinel (self-correcting rescale); P transposed via LDS.
// Output written bf16 into Ab[T][1024] for the final projection GEMM.
// ---------------------------------------------------------------------------
__global__ __launch_bounds__(32) void dilated_attn_kernel(
    const unsigned short* __restrict__ Qb,
    const unsigned short* __restrict__ Kb,
    const unsigned short* __restrict__ Vt,
    unsigned short* __restrict__ Ab)
{
    const int T = 2048;
    __shared__ unsigned short Pls[16 * 32];  // P tile, bf16, row-major 16x32

    int lane = threadIdx.x & 31;
    int hi = lane >> 4, lo = lane & 15;
    int h = blockIdx.y;
    int i0 = blockIdx.x * 16;

    v8f acc[4] = {};
    float mrow[8], lrow[8];
#pragma unroll
    for (int i = 0; i < 8; ++i) { mrow[i] = -1e30f; lrow[i] = 0.0f; }

    // Query A-fragments (reused across all key blocks): dh=64 -> 2 frags (K=32 each)
    const unsigned short* qrow = Qb + (size_t)(h * T + i0 + lo) * 64;
    v16bf aq0 = load_a_frag(qrow, hi);
    v16bf aq1 = load_a_frag(qrow + 32, hi);

    for (int jb = 0; jb < i0 + 16; jb += 32) {
        // ---- S = Q K^T for 16x32 tile (two 16x16 C tiles) ----
        v8f s[2] = {};
#pragma unroll
        for (int g = 0; g < 2; ++g) {
            const unsigned short* krow =
                Kb + (size_t)(h * T + jb + 16 * g + lo) * 64 + hi * 16;
            v16bf b0 = load_b_frag(krow);        // dh 0..31
            v16bf b1 = load_b_frag(krow + 32);   // dh 32..63
            s[g] = __builtin_amdgcn_wmma_f32_16x16x32_bf16(
                false, aq0, false, b0, (short)0, s[g], false, false);
            s[g] = __builtin_amdgcn_wmma_f32_16x16x32_bf16(
                false, aq1, false, b1, (short)0, s[g], false, false);
        }

        // ---- mask * count, online softmax ----
        float pv0[8], pv1[8];
#pragma unroll
        for (int i = 0; i < 8; ++i) {
            int ii = i0 + i + 8 * hi;

            int jj0 = jb + lo;
            int c0 = dil_count(ii, jj0, h);
            float v0 = (c0 > 0 && jj0 <= ii) ? s[0][i] * (float)c0 : -1e30f;

            int jj1 = jb + 16 + lo;
            int c1 = dil_count(ii, jj1, h);
            float v1 = (c1 > 0 && jj1 <= ii) ? s[1][i] * (float)c1 : -1e30f;

            float r = fmaxf(v0, v1);
#pragma unroll
            for (int off = 1; off <= 8; off <<= 1)
                r = fmaxf(r, __shfl_xor(r, off, 32));  // 16-lane row reduce

            float mn = fmaxf(mrow[i], r);
            float sc = __expf(mrow[i] - mn);
            float p0 = __expf(v0 - mn);
            float p1 = __expf(v1 - mn);
            float rs = p0 + p1;
#pragma unroll
            for (int off = 1; off <= 8; off <<= 1)
                rs += __shfl_xor(rs, off, 32);

            lrow[i] = lrow[i] * sc + rs;
            mrow[i] = mn;
#pragma unroll
            for (int dg = 0; dg < 4; ++dg) acc[dg][i] *= sc;
            pv0[i] = p0; pv1[i] = p1;
        }

        // ---- transpose P (C-layout -> A-layout) through LDS ----
#pragma unroll
        for (int i = 0; i < 8; ++i) {
            int M = i + 8 * hi;
            Pls[M * 32 + lo]      = f2bf(pv0[i]);
            Pls[M * 32 + 16 + lo] = f2bf(pv1[i]);
        }
        __syncthreads();
        Frag pf;
        {
            const unsigned short* pb = &Pls[lo * 32];
            pf.q[0] = *(const uint4*)(pb + hi * 16);
            pf.q[1] = *(const uint4*)(pb + hi * 16 + 8);
        }
        __syncthreads();

        // ---- acc += P @ V (keys = K dim) ----
#pragma unroll
        for (int dg = 0; dg < 4; ++dg) {
            const unsigned short* vp =
                Vt + (size_t)(h * 64 + dg * 16 + lo) * T + jb + hi * 16;
            v16bf bv = load_b_frag(vp);
            acc[dg] = __builtin_amdgcn_wmma_f32_16x16x32_bf16(
                false, pf.v, false, bv, (short)0, acc[dg], false, false);
        }
    }

    // ---- normalize, store bf16 into Ab[T][1024] ----
#pragma unroll
    for (int dg = 0; dg < 4; ++dg) {
#pragma unroll
        for (int i = 0; i < 8; ++i) {
            int t = i0 + i + 8 * hi;
            float o = acc[dg][i] / lrow[i];
            Ab[(size_t)t * 1024 + h * 64 + dg * 16 + lo] = f2bf(o);
        }
    }
}

// ---------------------------------------------------------------------------
// Launch
// ---------------------------------------------------------------------------
extern "C" void kernel_launch(void* const* d_in, const int* in_sizes, int n_in,
                              void* d_out, int out_size, void* d_ws, size_t ws_size,
                              hipStream_t stream) {
    const int T = 2048, D = 1024;

    const float* x  = (const float*)d_in[0];
    const float* Wq = (const float*)d_in[1];
    const float* bq = (const float*)d_in[2];
    const float* Wk = (const float*)d_in[3];
    const float* bk = (const float*)d_in[4];
    const float* Wv = (const float*)d_in[5];
    const float* bv = (const float*)d_in[6];
    const float* Wo = (const float*)d_in[7];
    const float* bo = (const float*)d_in[8];
    float* out = (float*)d_out;

    char* ws = (char*)d_ws;
    size_t off = 0;
    auto alloc = [&](size_t bytes) -> char* {
        char* p = ws + off;
        off += (bytes + 255) & ~(size_t)255;
        return p;
    };

    unsigned short* xb  = (unsigned short*)alloc((size_t)T * D * 2);
    unsigned short* Wqb = (unsigned short*)alloc((size_t)D * D * 2);
    unsigned short* Wkb = (unsigned short*)alloc((size_t)D * D * 2);
    unsigned short* Wvb = (unsigned short*)alloc((size_t)D * D * 2);
    unsigned short* Wob = (unsigned short*)alloc((size_t)D * D * 2);
    float* qf = (float*)alloc((size_t)T * D * 4);
    float* kf = (float*)alloc((size_t)T * D * 4);
    float* vf = (float*)alloc((size_t)T * D * 4);
    unsigned short* Qb  = (unsigned short*)alloc((size_t)T * D * 2);
    unsigned short* Kb  = (unsigned short*)alloc((size_t)T * D * 2);
    unsigned short* Vt  = (unsigned short*)alloc((size_t)T * D * 2);
    unsigned short* Ab  = (unsigned short*)alloc((size_t)T * D * 2);

    // 1) bf16 conversions
    f32_to_bf16_kernel<<<(T * D + 255) / 256, 256, 0, stream>>>(x, xb, T * D);
    f32_to_bf16_kernel<<<(D * D + 255) / 256, 256, 0, stream>>>(Wq, Wqb, D * D);
    f32_to_bf16_kernel<<<(D * D + 255) / 256, 256, 0, stream>>>(Wk, Wkb, D * D);
    f32_to_bf16_kernel<<<(D * D + 255) / 256, 256, 0, stream>>>(Wv, Wvb, D * D);
    f32_to_bf16_kernel<<<(D * D + 255) / 256, 256, 0, stream>>>(Wo, Wob, D * D);

    // 2) Q/K/V projections (fp32 out for RoPE precision)
    dim3 gg(D / 64, T / 64);  // (16, 32)
    gemm_bf16_wmma<<<gg, 128, 0, stream>>>(xb, Wqb, bq, qf, T, D, D);
    gemm_bf16_wmma<<<gg, 128, 0, stream>>>(xb, Wkb, bk, kf, T, D, D);
    gemm_bf16_wmma<<<gg, 128, 0, stream>>>(xb, Wvb, bv, vf, T, D, D);

    // 3) RoPE + pack to attention layouts
    rope_pack_kernel<<<T, 512, 0, stream>>>(qf, kf, vf, Qb, Kb, Vt);

    // 4) dilated attention (one wave per head x 16-query block)
    dilated_attn_kernel<<<dim3(T / 16, 16), 32, 0, stream>>>(Qb, Kb, Vt, Ab);

    // 5) output projection -> d_out (fp32)
    gemm_bf16_wmma<<<gg, 128, 0, stream>>>(Ab, Wob, bo, out, T, D, D);
}
